// ProjectedAdaptiveLogSoftmax_86517821214167
// MI455X (gfx1250) — compile-verified, matrix operand order checked by reading, not run on hardware
//
#include <hip/hip_runtime.h>
#include <hip/hip_bf16.h>
#include <math.h>
#include <stdint.h>

typedef __attribute__((ext_vector_type(16))) __bf16 v16bf;
typedef __attribute__((ext_vector_type(8)))  __bf16 v8bf;
typedef __attribute__((ext_vector_type(8)))  float  v8f;
typedef __attribute__((ext_vector_type(4)))  float  v4f;
typedef __attribute__((ext_vector_type(4)))  unsigned int v4u;
typedef __attribute__((ext_vector_type(8)))  int    v8i;
typedef __attribute__((ext_vector_type(4)))  int    v4i;

#define ROWS_PER_BLOCK 64
#define MT 4              // 16-row C tiles per block (64 rows)
#define WAVES 8
#define THREADS 256

__device__ __forceinline__ void online_update(float& m, float& s, float v) {
    float nm = fmaxf(m, v);
    s = s * __expf(m - nm) + __expf(v - nm);
    m = nm;
}
__device__ __forceinline__ void online_merge(float& m, float& s, float om, float os) {
    float nm = fmaxf(m, om);
    s = s * __expf(m - nm) + os * __expf(om - nm);
    m = nm;
}

// ---------------------------------------------------------------------------
// Projection GEMM: Hout[n][c] = (bf16) sum_k A[n][k] * B[k][c]
// A: [N][Kin] f32, B: [Kin][Nc] f32, Hout: [N][KpadOut] bf16.
// If Nc < KpadOut, columns [Nc, 2*Nc) are zero-filled (d=16 -> Kpad=32 case).
// ---------------------------------------------------------------------------
__global__ void proj_gemm_kernel(const float* __restrict__ A,
                                 const float* __restrict__ Bm,
                                 __bf16* __restrict__ Hout,
                                 int N, int Kin, int Nc, int KpadOut)
{
    const int lane   = threadIdx.x & 31;
    const int wave   = threadIdx.x >> 5;
    const int laneHi = lane >> 4;
    const int lane16 = lane & 15;
    const int rowBase = blockIdx.x * ROWS_PER_BLOCK;

    // Hoisted per-tile A row pointers.
    const float* ar[MT];
    #pragma unroll
    for (int t = 0; t < MT; ++t) {
        int row = rowBase + t * 16 + lane16;
        if (row >= N) row = N - 1;
        ar[t] = A + (size_t)row * Kin;
    }

    for (int cb = wave * 16; cb < Nc; cb += WAVES * 16) {
        const int myCol = cb + lane16;
        v8f acc[MT];
        #pragma unroll
        for (int t = 0; t < MT; ++t) acc[t] = (v8f)0.0f;

        for (int k0 = 0; k0 < Kin; k0 += 32) {
            const int base = k0 + (laneHi << 3);
            v16bf bf;
            #pragma unroll
            for (int i = 0; i < 8; ++i)
                bf[i] = (__bf16)Bm[(size_t)(base + i) * Nc + myCol];
            #pragma unroll
            for (int i = 0; i < 8; ++i)
                bf[8 + i] = (__bf16)Bm[(size_t)(base + 16 + i) * Nc + myCol];

            // Build ALL four A fragments first ...
            v16bf af[MT];
            #pragma unroll
            for (int t = 0; t < MT; ++t) {
                v4f a0 = *(const v4f*)(ar[t] + base);
                v4f a1 = *(const v4f*)(ar[t] + base + 4);
                v4f a2 = *(const v4f*)(ar[t] + base + 16);
                v4f a3 = *(const v4f*)(ar[t] + base + 20);
                #pragma unroll
                for (int i = 0; i < 4; ++i) {
                    af[t][i]      = (__bf16)a0[i];
                    af[t][4 + i]  = (__bf16)a1[i];
                    af[t][8 + i]  = (__bf16)a2[i];
                    af[t][12 + i] = (__bf16)a3[i];
                }
            }
            // ... then issue the four WMMAs back-to-back (no D->A/B hazards).
            #pragma unroll
            for (int t = 0; t < MT; ++t)
                acc[t] = __builtin_amdgcn_wmma_f32_16x16x32_bf16(
                    false, af[t], false, bf, (short)0, acc[t], false, false);
        }
        #pragma unroll
        for (int t = 0; t < MT; ++t) {
            #pragma unroll
            for (int r = 0; r < 8; ++r) {
                const int row = rowBase + t * 16 + r + (laneHi << 3);
                if (row < N) {
                    Hout[(size_t)row * KpadOut + myCol] = (__bf16)acc[t][r];
                    if (myCol + Nc < KpadOut)
                        Hout[(size_t)row * KpadOut + myCol + Nc] = (__bf16)0.0f;
                }
            }
        }
    }
}

// ---------------------------------------------------------------------------
// Fused GEMM + online log-sum-exp + logit capture.
// A-panel (64 rows x Kpad bf16) staged once per block into LDS via the
// Tensor Data Mover (padded rows for LDS bank spread), WMMA consumes from LDS.
// ---------------------------------------------------------------------------
__global__ void gemm_lse_kernel(const __bf16* __restrict__ A,
                                const float* __restrict__ W,
                                const float* __restrict__ W2,
                                const float* __restrict__ bias,
                                const float* __restrict__ bias2,
                                const int*   __restrict__ target,
                                float* __restrict__ lse_out,
                                float* __restrict__ tgt_out,
                                float* __restrict__ cluster_out,
                                int N, int Kpad, int dW,
                                int vocabMain, int vocabTotal,
                                int tOffset, int tClampMax,
                                int ldsStride,        // bytes per padded LDS row
                                int ishift, int padB, // reader pad compensation
                                int padIntervalCode, int padAmountCode)
{
    extern __shared__ __align__(16) char smem[];      // A panel (padded rows)
    __shared__ float redM[WAVES][MT][8][2];
    __shared__ float redS[WAVES][MT][8][2];

    const int lane   = threadIdx.x & 31;
    const int wave   = threadIdx.x >> 5;
    const int laneHi = lane >> 4;
    const int lane16 = lane & 15;
    const int rowBase = blockIdx.x * ROWS_PER_BLOCK;

    // ---- Stage A-panel into LDS ------------------------------------------
#if __has_builtin(__builtin_amdgcn_tensor_load_to_lds)
    if (wave == 0) {
        const uint32_t lds_base = (uint32_t)(uintptr_t)(void*)smem;
        const uint64_t gaddr = (uint64_t)(uintptr_t)A +
                               (uint64_t)rowBase * (uint64_t)(Kpad * 2);
        const uint32_t dim0 = (uint32_t)(Kpad >> 2);          // row in 8B units
        const uint32_t dim1 = (uint32_t)(N - rowBase);        // rows left (OOB->0)
        v4u g0 = (v4u)0u;
        g0.x = 1u;                                            // count=1, load, no gather
        g0.y = lds_base;                                      // lds_addr
        g0.z = (uint32_t)gaddr;                               // global_addr[31:0]
        g0.w = (uint32_t)((gaddr >> 32) & 0x1FFFFFFu) | (2u << 30); // [56:32] | type=2
        v8i g1 = (v8i)0;
        g1[0] = (3 << 16)                                     // data_size = 8B
              | (1 << 20)                                     // pad_enable
              | (padIntervalCode << 22)
              | (padAmountCode << 25);
        g1[1] = (int)((dim0 & 0xFFFFu) << 16);                // tensor_dim0[15:0]
        g1[2] = (int)(((dim0 >> 16) & 0xFFFFu) | ((dim1 & 0xFFFFu) << 16));
        g1[3] = (int)(((dim1 >> 16) & 0xFFFFu) | (dim0 << 16)); // tile_dim0 = dim0
        g1[4] = ROWS_PER_BLOCK;                               // tile_dim1 (tile_dim2=0)
        g1[5] = (int)dim0;                                    // tensor_dim0_stride lo
        g1[6] = 0;
        g1[7] = 0;
        v4i gz = (v4i)0;
#if __clang_major__ >= 23
        v8i gz8 = (v8i)0;
        __builtin_amdgcn_tensor_load_to_lds(g0, g1, gz, gz, gz8, 0);
#else
        __builtin_amdgcn_tensor_load_to_lds(g0, g1, gz, gz, 0);
#endif
        __builtin_amdgcn_s_wait_tensorcnt((short)0);
    }
#else
    // Fallback: cooperative copy with identical padded layout.
    {
        const int rowBytes = Kpad * 2;
        const int chPerRow = rowBytes >> 4;
        const int chunks = ROWS_PER_BLOCK * chPerRow;
        for (int c = threadIdx.x; c < chunks; c += THREADS) {
            const int row = c / chPerRow, j = c % chPerRow;
            const int gRow = rowBase + row;
            v8bf val;
            for (int i = 0; i < 8; ++i) val[i] = (__bf16)0.0f;
            if (gRow < N) val = *(const v8bf*)(A + (size_t)gRow * Kpad + j * 8);
            const int kb = j * 16;
            *(v8bf*)(smem + row * ldsStride + kb + ((kb >> ishift) * padB)) = val;
        }
    }
#endif
    __syncthreads();

    // Pre-clamp targets for this lane's C-register rows.
    int tcl[MT][8];
    #pragma unroll
    for (int t = 0; t < MT; ++t)
        #pragma unroll
        for (int r = 0; r < 8; ++r) {
            int row = rowBase + t * 16 + r + (laneHi << 3);
            if (row >= N) row = N - 1;
            int tv = target[row] - tOffset;
            tv = tv < 0 ? 0 : (tv > tClampMax ? tClampMax : tv);
            tcl[t][r] = tv;
        }

    float runM[MT][8], runS[MT][8];
    #pragma unroll
    for (int t = 0; t < MT; ++t)
        #pragma unroll
        for (int r = 0; r < 8; ++r) { runM[t][r] = -__builtin_inff(); runS[t][r] = 0.0f; }

    // Hoisted per-tile LDS row pointers (A-fragment rows: M = lane16).
    const char* rp[MT];
    #pragma unroll
    for (int t = 0; t < MT; ++t)
        rp[t] = smem + (size_t)(t * 16 + lane16) * ldsStride;

    for (int cb = wave * 16; cb < vocabTotal; cb += WAVES * 16) {
        const int  myCol    = cb + lane16;
        const bool colValid = (myCol < vocabTotal);
        const int  wr = colValid ? myCol : (vocabTotal - 1);
        const float* wrow = (wr < vocabMain)
                                ? (W  + (size_t)wr * dW)
                                : (W2 + (size_t)(wr - vocabMain) * dW);

        v8f acc[MT];
        #pragma unroll
        for (int t = 0; t < MT; ++t) acc[t] = (v8f)0.0f;

        for (int k0 = 0; k0 < Kpad; k0 += 32) {
            const int base = k0 + (laneHi << 3);
            v16bf bf;
            if (base < dW) {
                v4f f0 = *(const v4f*)(wrow + base);
                v4f f1 = *(const v4f*)(wrow + base + 4);
                #pragma unroll
                for (int i = 0; i < 4; ++i) { bf[i] = (__bf16)f0[i]; bf[4 + i] = (__bf16)f1[i]; }
            } else {
                #pragma unroll
                for (int i = 0; i < 8; ++i) bf[i] = (__bf16)0.0f;
            }
            if (base + 16 < dW) {
                v4f f2 = *(const v4f*)(wrow + base + 16);
                v4f f3 = *(const v4f*)(wrow + base + 20);
                #pragma unroll
                for (int i = 0; i < 4; ++i) { bf[8 + i] = (__bf16)f2[i]; bf[12 + i] = (__bf16)f3[i]; }
            } else {
                #pragma unroll
                for (int i = 0; i < 8; ++i) bf[8 + i] = (__bf16)0.0f;
            }

            // A fragments from LDS (pad-compensated byte offsets): load ALL
            // four first, then issue the four WMMAs back-to-back.
            const int kb0 = base * 2;
            const int kb1 = base * 2 + 32;
            const int o0 = kb0 + ((kb0 >> ishift) * padB);
            const int o1 = kb1 + ((kb1 >> ishift) * padB);
            v16bf af[MT];
            #pragma unroll
            for (int t = 0; t < MT; ++t) {
                v8bf a0 = *(const v8bf*)(rp[t] + o0);
                v8bf a1 = *(const v8bf*)(rp[t] + o1);
                #pragma unroll
                for (int i = 0; i < 8; ++i) { af[t][i] = a0[i]; af[t][8 + i] = a1[i]; }
            }
            #pragma unroll
            for (int t = 0; t < MT; ++t)
                acc[t] = __builtin_amdgcn_wmma_f32_16x16x32_bf16(
                    false, af[t], false, bf, (short)0, acc[t], false, false);
        }

        float bv = 0.0f;
        if (colValid)
            bv = (myCol < vocabMain) ? bias[myCol] : bias2[myCol - vocabMain];

        #pragma unroll
        for (int t = 0; t < MT; ++t)
            #pragma unroll
            for (int r = 0; r < 8; ++r) {
                const int row = rowBase + t * 16 + r + (laneHi << 3);
                const float v = acc[t][r] + bv;
                if (colValid) {
                    online_update(runM[t][r], runS[t][r], v);
                    if (row < N) {
                        if (myCol == tcl[t][r]) tgt_out[row] = v;
                        if (cluster_out && myCol >= vocabMain)
                            cluster_out[(size_t)row * 3 + (myCol - vocabMain)] = v;
                    }
                }
            }
    }

    // Butterfly over the 16 column-lanes of each half-wave (masks stay in-half).
    #pragma unroll
    for (int t = 0; t < MT; ++t)
        #pragma unroll
        for (int r = 0; r < 8; ++r) {
            float m = runM[t][r], s = runS[t][r];
            #pragma unroll
            for (int off = 1; off < 16; off <<= 1) {
                float om = __shfl_xor(m, off, 32);
                float os = __shfl_xor(s, off, 32);
                online_merge(m, s, om, os);
            }
            if (lane16 == 0) { redM[wave][t][r][laneHi] = m; redS[wave][t][r][laneHi] = s; }
        }
    __syncthreads();

    if ((int)threadIdx.x < ROWS_PER_BLOCK) {
        const int j = threadIdx.x;
        const int t = j >> 4, rem = j & 15, hi = rem >> 3, r = rem & 7;
        float M = -__builtin_inff(), S = 0.0f;
        for (int w = 0; w < WAVES; ++w)
            online_merge(M, S, redM[w][t][r][hi], redS[w][t][r][hi]);
        const int row = rowBase + j;
        if (row < N) lse_out[row] = M + logf(S);
    }
}

// ---------------------------------------------------------------------------
__global__ void finalize_kernel(const int* __restrict__ target,
                                const float* __restrict__ lse0, const float* __restrict__ tgt0,
                                const float* __restrict__ cluster,
                                const float* __restrict__ lse1, const float* __restrict__ tgt1,
                                const float* __restrict__ lse2, const float* __restrict__ tgt2,
                                const float* __restrict__ lse3, const float* __restrict__ tgt3,
                                float* __restrict__ out, int N)
{
    const int n = blockIdx.x * blockDim.x + threadIdx.x;
    if (n >= N) return;
    const int t = target[n];
    const float l0 = lse0[n];
    float nll = -(tgt0[n] - l0);
    if (t >= 20000 && t < 30000) nll = -((cluster[(size_t)n * 3 + 2] - l0) + (tgt1[n] - lse1[n]));
    if (t >= 30000 && t < 40000) nll = -((cluster[(size_t)n * 3 + 1] - l0) + (tgt2[n] - lse2[n]));
    if (t >= 40000)              nll = -((cluster[(size_t)n * 3 + 0] - l0) + (tgt3[n] - lse3[n]));
    out[n] = nll;
}

// ---------------------------------------------------------------------------
extern "C" void kernel_launch(void* const* d_in, const int* in_sizes, int n_in,
                              void* d_out, int out_size, void* d_ws, size_t ws_size,
                              hipStream_t stream) {
    const float* hidden    = (const float*)d_in[0];
    const int*   target    = (const int*)  d_in[1];
    const float* proj0     = (const float*)d_in[2];
    const float* w0        = (const float*)d_in[3];
    const float* b0        = (const float*)d_in[4];
    const float* proj1     = (const float*)d_in[5];
    const float* w1        = (const float*)d_in[6];
    const float* b1        = (const float*)d_in[7];
    const float* proj2     = (const float*)d_in[8];
    const float* w2        = (const float*)d_in[9];
    const float* b2        = (const float*)d_in[10];
    const float* proj3     = (const float*)d_in[11];
    const float* w3        = (const float*)d_in[12];
    const float* b3        = (const float*)d_in[13];
    const float* cluster_w = (const float*)d_in[14];
    const float* cluster_b = (const float*)d_in[15];

    const int N = in_sizes[1];          // 4096

    size_t off = 0;
    char* base = (char*)d_ws;
    auto carve = [&](size_t bytes) -> void* {
        void* p = base + off;
        off = (off + bytes + 255) & ~(size_t)255;
        return p;
    };
    __bf16* h0 = (__bf16*)carve((size_t)N * 1024 * 2);
    __bf16* h1 = (__bf16*)carve((size_t)N * 256  * 2);
    __bf16* h2 = (__bf16*)carve((size_t)N * 64   * 2);
    __bf16* h3 = (__bf16*)carve((size_t)N * 32   * 2);
    float* lse0 = (float*)carve((size_t)N * 4);
    float* lse1 = (float*)carve((size_t)N * 4);
    float* lse2 = (float*)carve((size_t)N * 4);
    float* lse3 = (float*)carve((size_t)N * 4);
    float* tgt0 = (float*)carve((size_t)N * 4);
    float* tgt1 = (float*)carve((size_t)N * 4);
    float* tgt2 = (float*)carve((size_t)N * 4);
    float* tgt3 = (float*)carve((size_t)N * 4);
    float* clst = (float*)carve((size_t)N * 3 * 4);

    dim3 grid((N + ROWS_PER_BLOCK - 1) / ROWS_PER_BLOCK);
    dim3 block(THREADS);

    // Stage A: projections -> bf16 panels.
    proj_gemm_kernel<<<grid, block, 0, stream>>>(hidden, proj0, h0, N, 1024, 1024, 1024);
    proj_gemm_kernel<<<grid, block, 0, stream>>>(hidden, proj1, h1, N, 1024, 256,  256);
    proj_gemm_kernel<<<grid, block, 0, stream>>>(hidden, proj2, h2, N, 1024, 64,   64);
    proj_gemm_kernel<<<grid, block, 0, stream>>>(hidden, proj3, h3, N, 1024, 16,   32);

    // LDS row strides (data + TDM pad): head pads 16B per 1024B (2x/row),
    // tails pad 16B once per row (conflict-free, 16B-aligned).
    const int strideHead = 1024 * 2 + 32;   // 2080
    const int strideT1   = 256 * 2 + 16;    // 528
    const int strideT2   = 64 * 2 + 16;     // 144
    const int strideT3   = 32 * 2 + 16;     // 80

    // Stage B: head (20000 vocab + 3 cluster cols) with online LSE + captures.
    gemm_lse_kernel<<<grid, block, ROWS_PER_BLOCK * strideHead, stream>>>(
        h0, w0, cluster_w, b0, cluster_b, target, lse0, tgt0, clst,
        N, 1024, 1024, 20000, 20003, 0, 19999,
        strideHead, /*ishift*/10, /*padB*/16, /*intervalCode(1024B)*/7, /*amount(4dw)*/3);
    gemm_lse_kernel<<<grid, block, ROWS_PER_BLOCK * strideT1, stream>>>(
        h1, w1, nullptr, b1, nullptr, target, lse1, tgt1, nullptr,
        N, 256, 256, 10000, 10000, 20000, 9999,
        strideT1, 30, 0, /*512B row*/6, 3);
    gemm_lse_kernel<<<grid, block, ROWS_PER_BLOCK * strideT2, stream>>>(
        h2, w2, nullptr, b2, nullptr, target, lse2, tgt2, nullptr,
        N, 64, 64, 10000, 10000, 30000, 9999,
        strideT2, 30, 0, /*128B row*/4, 3);
    gemm_lse_kernel<<<grid, block, ROWS_PER_BLOCK * strideT3, stream>>>(
        h3, w3, nullptr, b3, nullptr, target, lse3, tgt3, nullptr,
        N, 32, 16, 10000, 10000, 40000, 9999,
        strideT3, 30, 0, /*64B row*/3, 3);

    // Stage C: assemble NLL.
    finalize_kernel<<<dim3((N + 255) / 256), dim3(256), 0, stream>>>(
        target, lse0, tgt0, clst, lse1, tgt1, lse2, tgt2, lse3, tgt3,
        (float*)d_out, N);
}
